// ACDSS_80676665688693
// MI455X (gfx1250) — matrix operand order repaired and checked by reading, not compiled, not run on hardware
//
#include <hip/hip_runtime.h>
#include <math.h>

// ---------------------------------------------------------------------------
// CDNA5 (gfx1250, wave32) fused NRI forward pass.
// Heavy GEMMs: v_wmma_f32_16x16x32_bf16, k-outer loop, 512-thread blocks
// (16 waves = 8 M-tiles x 2 N-halves, 8 accumulators/wave -> no spills),
// weight slices staged coalesced into LDS, global_prefetch of the next slice.
// ---------------------------------------------------------------------------

typedef __attribute__((ext_vector_type(16))) __bf16 bf16x16;
typedef __attribute__((ext_vector_type(8)))  float  floatx8;

union Frag { bf16x16 v; uint4 q[2]; };

__device__ __forceinline__ unsigned short f2bf(float f) {
  unsigned u = __float_as_uint(f);
  unsigned r = u + 0x7FFFu + ((u >> 16) & 1u);   // round-to-nearest-even
  return (unsigned short)(r >> 16);
}
__device__ __forceinline__ float bf2f(unsigned short h) {
  return __uint_as_float(((unsigned)h) << 16);
}
__device__ __forceinline__ float eluf(float v)     { return v > 0.f ? v : (__expf(v) - 1.f); }
__device__ __forceinline__ float sigmoidf_(float v){ return 1.f / (1.f + __expf(-v)); }
__device__ __forceinline__ floatx8 fzero8() { floatx8 z; for (int i = 0; i < 8; ++i) z[i] = 0.f; return z; }

__device__ __forceinline__ floatx8 wmma_bf16(bf16x16 a, bf16x16 b, floatx8 c) {
  return __builtin_amdgcn_wmma_f32_16x16x32_bf16(false, a, false, b, (short)0, c, false, false);
}

// A-operand fragment (16x32 bf16): lane holds row m0+(lane&15); two 16B chunks.
__device__ __forceinline__ bf16x16 load_afrag(const unsigned short* rowk0, int hi) {
  Frag f;
  f.q[0] = *(const uint4*)(rowk0 + 8 * hi);
  f.q[1] = *(const uint4*)(rowk0 + 16 + 8 * hi);
  return f.v;
}
// B-operand fragment (32x16 bf16): lane holds column n (w points at row k0, col n), k in table order, row stride 256.
__device__ __forceinline__ bf16x16 load_bfrag(const unsigned short* w, int hi) {
  Frag f;
  unsigned* u = (unsigned*)f.q;
#pragma unroll
  for (int i = 0; i < 8; ++i) {
    int s = 2 * i;
    int k = (s < 8) ? (8 * hi + s) : (16 + 8 * hi + (s - 8));
    u[i] = (unsigned)w[k * 256] | ((unsigned)w[(k + 1) * 256] << 16);
  }
  return f.v;
}

// ---------------------------------------------------------------------------
// Block-cooperative GEMM layer (512 threads = 16 waves).
// Wave (mt, nh): M-tile mt (16 rows), N-half nh (8 n-tiles), acc[8].
// Per 32-wide k-step: stage W[k0..k0+64)[0..256) into LDS coalesced (32KB),
// prefetch next slice, then each wave issues 8 WMMAs re-using one A fragment.
// arow: &A[lane-row][0] (global or LDS), row stride == K elements.
// ---------------------------------------------------------------------------
__device__ __forceinline__ void gemm_layer(const unsigned short* arow, int K,
                                           const unsigned short* __restrict__ W,
                                           unsigned short* wsl,        // LDS 32*256 u16
                                           floatx8* acc,
                                           int tid, int lane, int hi, int nh) {
  for (int k0 = 0; k0 < K; k0 += 32) {
    {
      const uint4* src = (const uint4*)(W + (size_t)k0 * 256);
      uint4* dst = (uint4*)wsl;
#pragma unroll
      for (int i = 0; i < 4; ++i) dst[tid + 512 * i] = src[tid + 512 * i];
    }
    if (k0 + 32 < K)
      __builtin_prefetch(W + (size_t)(k0 + 32) * 256 + tid * 16, 0, 0);
    __syncthreads();
    bf16x16 fa = load_afrag(arow + k0, hi);
    const unsigned short* wl = wsl + (lane & 15) + nh * 128;
#pragma unroll
    for (int nt = 0; nt < 8; ++nt) {
      bf16x16 fb = load_bfrag(wl + nt * 16, hi);
      acc[nt] = wmma_bf16(fa, fb, acc[nt]);
    }
    __syncthreads();
  }
}

__device__ __forceinline__ void acc_zero8(floatx8* acc) {
#pragma unroll
  for (int nt = 0; nt < 8; ++nt) acc[nt] = fzero8();
}

// acc -> ELU(+bias) -> bf16 into LDS activation rows.
__device__ __forceinline__ void epilogue_elu(floatx8* acc, const float* __restrict__ B,
                                             unsigned short* act, int m0, int lo, int hi, int nh) {
#pragma unroll
  for (int nt = 0; nt < 8; ++nt) {
    int n = (nh * 8 + nt) * 16 + lo;
    float bias = B[n];
#pragma unroll
    for (int r = 0; r < 8; ++r)
      act[(m0 + r + 8 * hi) * 256 + n] = f2bf(eluf(acc[nt][r] + bias));
  }
}

// ---------------------------------------------------------------------------
// Utility kernels
// ---------------------------------------------------------------------------
__global__ void k_zero_f32(float* p, int n) {
  int i = blockIdx.x * 256 + threadIdx.x;
  if (i < n) p[i] = 0.f;
}

// f32 -> bf16 with optional zero row padding (rows of 256 elems).
__global__ void k_f2bf_pad(const float* __restrict__ src, unsigned short* __restrict__ dst,
                           int srcRows, int n) {
  int i = blockIdx.x * 256 + threadIdx.x;
  if (i < n) {
    int row = i >> 8;
    dst[i] = (row < srcRows) ? f2bf(src[i]) : (unsigned short)0;
  }
}

// Training-mode BatchNorm1d over rows (biased var), output bf16.
__global__ void k_bn(const float* __restrict__ in, int M,
                     const float* __restrict__ gamma, const float* __restrict__ beta,
                     unsigned short* __restrict__ out) {
  __shared__ float red[256];
  const int c = blockIdx.x, tid = threadIdx.x;
  float s = 0.f, s2 = 0.f;
  for (int r = tid; r < M; r += 256) { float v = in[(size_t)r * 256 + c]; s += v; s2 += v * v; }
  red[tid] = s; __syncthreads();
  for (int off = 128; off > 0; off >>= 1) { if (tid < off) red[tid] += red[tid + off]; __syncthreads(); }
  float tot = red[0]; __syncthreads();
  red[tid] = s2; __syncthreads();
  for (int off = 128; off > 0; off >>= 1) { if (tid < off) red[tid] += red[tid + off]; __syncthreads(); }
  float tot2 = red[0];
  float mean = tot / M;
  float var  = tot2 / M - mean * mean;
  float inv  = rsqrtf(var + 1e-5f) * gamma[c];
  float be   = beta[c];
  for (int r = tid; r < M; r += 256)
    out[(size_t)r * 256 + c] = f2bf((in[(size_t)r * 256 + c] - mean) * inv + be);
}

// node2edge gather: A2[b*992+e] = [z1[b,s(e)], z1[b,r(e)]]
__global__ void k_gather2(const unsigned short* __restrict__ z1, unsigned short* __restrict__ A2) {
  int row = blockIdx.x;
  int b = row / 992, e = row % 992;
  int r = e / 31, j = e % 31, s = j + (j >= r ? 1 : 0);
  int c = threadIdx.x;
  A2[(size_t)row * 512 + c]       = z1[(size_t)(b * 32 + s) * 256 + c];
  A2[(size_t)row * 512 + 256 + c] = z1[(size_t)(b * 32 + r) * 256 + c];
}

// edge2node mean: A3[b,n] = (1/E) * sum_{incoming edges} z2
__global__ void k_edge2node(const unsigned short* __restrict__ z2, unsigned short* __restrict__ A3) {
  int bn = blockIdx.x;            // b*32 + n
  int b = bn >> 5, n = bn & 31;
  int c = threadIdx.x;
  float s = 0.f;
  for (int k = 0; k < 31; ++k) s += bf2f(z2[(size_t)(b * 992 + n * 31 + k) * 256 + c]);
  A3[(size_t)bn * 256 + c] = f2bf(s * (1.f / 992.f));
}

// A4[b*992+e] = [z3[b,s], z3[b,r], skip=z2[b,e]]
__global__ void k_gather3(const unsigned short* __restrict__ z3, const unsigned short* __restrict__ z2,
                          unsigned short* __restrict__ A4) {
  int row = blockIdx.x;
  int b = row / 992, e = row % 992;
  int r = e / 31, j = e % 31, s = j + (j >= r ? 1 : 0);
  int c = threadIdx.x;
  A4[(size_t)row * 768 + c]       = z3[(size_t)(b * 32 + s) * 256 + c];
  A4[(size_t)row * 768 + 256 + c] = z3[(size_t)(b * 32 + r) * 256 + c];
  A4[(size_t)row * 768 + 512 + c] = z2[(size_t)row * 256 + c];
}

// logits = z4 @ W[256x4] + b ; softmax over 4
__global__ void k_enc_out(const unsigned short* __restrict__ z4,
                          const float* __restrict__ W, const float* __restrict__ B,
                          float* __restrict__ edges) {
  int row = blockIdx.x * 256 + threadIdx.x;
  if (row >= 3968) return;
  float l0 = B[0], l1 = B[1], l2 = B[2], l3 = B[3];
  const unsigned short* zr = z4 + (size_t)row * 256;
  for (int c = 0; c < 256; ++c) {
    float z = bf2f(zr[c]);
    l0 += z * W[c * 4 + 0]; l1 += z * W[c * 4 + 1];
    l2 += z * W[c * 4 + 2]; l3 += z * W[c * 4 + 3];
  }
  float mx = fmaxf(fmaxf(l0, l1), fmaxf(l2, l3));
  float e0 = __expf(l0 - mx), e1 = __expf(l1 - mx), e2 = __expf(l2 - mx), e3 = __expf(l3 - mx);
  float inv = 1.f / (e0 + e1 + e2 + e3);
  float* o = edges + (size_t)row * 4;
  o[0] = e0 * inv; o[1] = e1 * inv; o[2] = e2 * inv; o[3] = e3 * inv;
}

// aug[row] = [x, agg(256), pad..288] as bf16
__global__ void k_build_aug(const float* __restrict__ x, const float* __restrict__ agg,
                            unsigned short* __restrict__ aug) {
  const size_t row = blockIdx.x;
  for (int c = threadIdx.x; c < 288; c += 256) {
    float v;
    if (c == 0) v = x[row];
    else if (c <= 256) v = agg[row * 256 + (c - 1)];
    else v = 0.f;
    aug[row * 288 + c] = f2bf(v);
  }
}

// ---------------------------------------------------------------------------
// Generic fused 3-layer MLP (ELU, ELU, linear): A[M][K] bf16 -> out[M][256] f32
// grid = M/128 blocks x 512 threads (16 waves: 8 M-tiles x 2 N-halves).
// ---------------------------------------------------------------------------
__global__ void k_fused_mlp3(const unsigned short* __restrict__ A, int K,
                             const unsigned short* __restrict__ W1, const float* __restrict__ B1,
                             const unsigned short* __restrict__ W2, const float* __restrict__ B2,
                             const unsigned short* __restrict__ W3, const float* __restrict__ B3,
                             float* __restrict__ out) {
  extern __shared__ unsigned short sm_mlp[];
  unsigned short* act1 = sm_mlp;                 // 128*256 bf16
  unsigned short* act2 = sm_mlp + 128 * 256;     // 128*256 bf16
  unsigned short* wsl  = sm_mlp + 2 * 128 * 256; // 32*256 bf16 staging
  const int tid = threadIdx.x;
  const int wave = tid >> 5, lane = tid & 31;
  const int lo = lane & 15, hi = lane >> 4;
  const int m0 = (wave >> 1) * 16;
  const int nh = wave & 1;
  const int rowBase = blockIdx.x * 128;
  floatx8 acc[8];

  // layer 1 (A from global)
  acc_zero8(acc);
  gemm_layer(A + (size_t)(rowBase + m0 + lo) * K, K, W1, wsl, acc, tid, lane, hi, nh);
  epilogue_elu(acc, B1, act1, m0, lo, hi, nh);

  // layer 2 (LDS -> LDS); gemm_layer's internal barriers order act writes/reads.
  acc_zero8(acc);
  gemm_layer(act1 + (m0 + lo) * 256, 256, W2, wsl, acc, tid, lane, hi, nh);
  epilogue_elu(acc, B2, act2, m0, lo, hi, nh);

  // layer 3 (LDS -> global f32, no activation)
  acc_zero8(acc);
  gemm_layer(act2 + (m0 + lo) * 256, 256, W3, wsl, acc, tid, lane, hi, nh);
#pragma unroll
  for (int nt = 0; nt < 8; ++nt) {
    int n = (nh * 8 + nt) * 16 + lo;
    float bias = B3[n];
#pragma unroll
    for (int r = 0; r < 8; ++r)
      out[(size_t)(rowBase + m0 + r + 8 * hi) * 256 + n] = acc[nt][r] + bias;
  }
}

// ---------------------------------------------------------------------------
// One GRU-style RNN step. grid = 8 blocks x 256 threads; block = 16 rows.
// (Small weights, L2-resident; 2-3 accumulators, no spill risk.)
// ---------------------------------------------------------------------------
__global__ void k_rnn_step(const float* __restrict__ x, int t,
                           const float* __restrict__ h_in, float* __restrict__ h_out,
                           const float* __restrict__ inpW, const float* __restrict__ inpB,
                           const unsigned short* __restrict__ Wg1, const float* __restrict__ Bg1,
                           const unsigned short* __restrict__ Wg2, const float* __restrict__ Bg2,
                           const unsigned short* __restrict__ Wo,  const float* __restrict__ Bo,
                           const int* __restrict__ n_steps, float* __restrict__ node_h) {
  __shared__ unsigned short A[16 * 512];   // [xi, h] then [xi, g1*h]
  __shared__ float hbuf[16 * 256];
  const int tid = threadIdx.x;
  const int wave = tid >> 5, lane = tid & 31, lo = lane & 15, hi = lane >> 4;
  const int r0 = blockIdx.x * 16;

  for (int i = tid; i < 16 * 256; i += 256) {
    int mr = i >> 8, c = i & 255;
    int row = r0 + mr, b = row >> 5, n = row & 31;
    float xv = x[(b * 64 + t) * 32 + n];
    A[mr * 512 + c] = f2bf(xv * inpW[c] + inpB[c]);
    float h = h_in[(size_t)row * 256 + c];
    hbuf[i] = h;
    A[mr * 512 + 256 + c] = f2bf(h);
  }
  __syncthreads();

  float g1v[2][8], g2v[2][8];
  {
    const unsigned short* arow = A + lo * 512;
    for (int j = 0; j < 2; ++j) {
      int nt = wave * 2 + j;
      floatx8 a1 = fzero8(), a2 = fzero8();
      for (int k0 = 0; k0 < 512; k0 += 32) {
        bf16x16 fa  = load_afrag(arow + k0, hi);
        bf16x16 fb1 = load_bfrag(Wg1 + (size_t)k0 * 256 + nt * 16 + lo, hi);
        bf16x16 fb2 = load_bfrag(Wg2 + (size_t)k0 * 256 + nt * 16 + lo, hi);
        a1 = wmma_bf16(fa, fb1, a1);
        a2 = wmma_bf16(fa, fb2, a2);
      }
      int n = nt * 16 + lo;
      float bb1 = Bg1[n], bb2 = Bg2[n];
#pragma unroll
      for (int r = 0; r < 8; ++r) {
        g1v[j][r] = sigmoidf_(a1[r] + bb1);
        g2v[j][r] = sigmoidf_(a2[r] + bb2);
      }
    }
  }
  __syncthreads();
  for (int j = 0; j < 2; ++j) {   // overwrite h-half with g1*h
    int nt = wave * 2 + j;
    int n = nt * 16 + lo;
#pragma unroll
    for (int r = 0; r < 8; ++r) {
      int m = r + 8 * hi;
      A[m * 512 + 256 + n] = f2bf(g1v[j][r] * hbuf[m * 256 + n]);
    }
  }
  __syncthreads();
  {
    const unsigned short* arow = A + lo * 512;
    for (int j = 0; j < 2; ++j) {
      int nt = wave * 2 + j;
      floatx8 acc = fzero8();
      for (int k0 = 0; k0 < 512; k0 += 32) {
        bf16x16 fa = load_afrag(arow + k0, hi);
        bf16x16 fb = load_bfrag(Wo + (size_t)k0 * 256 + nt * 16 + lo, hi);
        acc = wmma_bf16(fa, fb, acc);
      }
      int n = nt * 16 + lo;
      float bb = Bo[n];
#pragma unroll
      for (int r = 0; r < 8; ++r) {
        int m = r + 8 * hi;
        float hn = tanhf(acc[r] + bb);
        float g2 = g2v[j][r];
        float h  = (1.f - g2) * hbuf[m * 256 + n] + g2 * hn;
        int row  = r0 + m;
        h_out[(size_t)row * 256 + n] = h;
        int b  = row >> 5;
        int ns = n_steps[b];
        int tsel = (ns < 1 ? 1 : (ns > 64 ? 64 : ns)) - 1;
        if (t == tsel) node_h[(size_t)row * 256 + n] = h;
      }
    }
  }
}

// ---------------------------------------------------------------------------
// Decoder message kernel. grid = B*T*(N/4) = 2048 blocks x 512 threads;
// block = 4 receivers x 31 incoming edges = 124 rows (padded to 128).
// Loops 3 edge types: layer1(K=2, VALU) + 2 WMMA layers + weighted
// accumulation + receiver-group reduce (no atomics).
// ---------------------------------------------------------------------------
struct MsgW {
  const float* w1; const float* b1;
  const unsigned short* w2; const float* b2;
  const unsigned short* w3; const float* b3;
};

__global__ void k_dec_msg(const float* __restrict__ x, const float* __restrict__ edges,
                          MsgW ty0, MsgW ty1, MsgW ty2, float* __restrict__ agg) {
  extern __shared__ char smraw[];
  unsigned short* act1 = (unsigned short*)smraw;       // 128*256 bf16
  unsigned short* act2 = act1 + 128 * 256;             // 128*256 bf16
  unsigned short* wsl  = act2 + 128 * 256;             // 32*256 bf16 staging
  float* accs = (float*)(wsl + 32 * 256);              // 128*256 f32
  float* sxl  = accs + 128 * 256;                      // 128
  float* rxl  = sxl + 128;                             // 128
  float* wel  = rxl + 128;                             // 3*128

  const int tid = threadIdx.x;
  const int wave = tid >> 5, lane = tid & 31, lo = lane & 15, hi = lane >> 4;
  const int m0 = (wave >> 1) * 16;
  const int nh = wave & 1;
  const int g = blockIdx.x;
  const int n0 = (g & 7) * 4;
  const int bt = g >> 3;
  const int t = bt & 63, b = bt >> 6;
  const float* xt = x + (size_t)(b * 64 + t) * 32;

  if (tid < 128) {
    int i = tid;
    float sx = 0.f, rx = 0.f, w1 = 0.f, w2 = 0.f, w3 = 0.f;
    if (i < 124) {
      int e = n0 * 31 + i;
      int r = e / 31, j = e % 31;
      int s = j + (j >= r ? 1 : 0);
      sx = xt[s]; rx = xt[r];
      const float* ed = edges + (size_t)(b * 992 + e) * 4;
      w1 = ed[1]; w2 = ed[2]; w3 = ed[3];
    }
    sxl[i] = sx; rxl[i] = rx;
    wel[0 * 128 + i] = w1; wel[1 * 128 + i] = w2; wel[2 * 128 + i] = w3;
  }
  for (int i = tid; i < 128 * 256; i += 512) accs[i] = 0.f;
  __syncthreads();

  floatx8 acc[8];
  MsgW Ws[3] = { ty0, ty1, ty2 };
  for (int ty = 0; ty < 3; ++ty) {
    MsgW W = Ws[ty];
    // layer 1: K=2 -> VALU
    for (int i = tid; i < 128 * 256; i += 512) {
      int mr = i >> 8, c = i & 255;
      float v = sxl[mr] * W.w1[c] + rxl[mr] * W.w1[256 + c] + W.b1[c];
      act1[i] = f2bf(eluf(v));
    }
    __syncthreads();
    // layer 2
    acc_zero8(acc);
    gemm_layer(act1 + (m0 + lo) * 256, 256, W.w2, wsl, acc, tid, lane, hi, nh);
    epilogue_elu(acc, W.b2, act2, m0, lo, hi, nh);
    // layer 3 + weighted accumulate
    acc_zero8(acc);
    gemm_layer(act2 + (m0 + lo) * 256, 256, W.w3, wsl, acc, tid, lane, hi, nh);
#pragma unroll
    for (int nt = 0; nt < 8; ++nt) {
      int n = (nh * 8 + nt) * 16 + lo;
      float bias = W.b3[n];
#pragma unroll
      for (int r = 0; r < 8; ++r) {
        int m = m0 + r + 8 * hi;
        accs[m * 256 + n] += wel[ty * 128 + m] * (acc[nt][r] + bias);
      }
    }
    __syncthreads();
  }
  // reduce 31-row receiver groups -> agg
  for (int i = tid; i < 4 * 256; i += 512) {
    int jj = i >> 8, c = i & 255;
    float s = 0.f;
    for (int k = 0; k < 31; ++k) s += accs[(jj * 31 + k) * 256 + c];
    agg[((size_t)((b * 64 + t) * 32 + n0 + jj)) * 256 + c] = s;
  }
}

// ---------------------------------------------------------------------------
// Decoder output MLP: aug[8192][288] -> 256 -> 256 -> 1 ; pred = x + out.
// grid = 64 blocks x 512 threads.
// ---------------------------------------------------------------------------
__global__ void k_dec_out(const unsigned short* __restrict__ aug,
                          const unsigned short* __restrict__ W1, const float* __restrict__ B1,
                          const unsigned short* __restrict__ W2, const float* __restrict__ B2,
                          const float* __restrict__ W3, const float* __restrict__ B3,
                          const float* __restrict__ x, float* __restrict__ pred) {
  extern __shared__ unsigned short sm_do[];
  unsigned short* act1 = sm_do;                 // 128*256
  unsigned short* act2 = sm_do + 128 * 256;     // 128*256
  unsigned short* wsl  = sm_do + 2 * 128 * 256; // 32*256 staging
  const int tid = threadIdx.x;
  const int wave = tid >> 5, lane = tid & 31, lo = lane & 15, hi = lane >> 4;
  const int m0 = (wave >> 1) * 16;
  const int nh = wave & 1;
  const int rowBase = blockIdx.x * 128;
  floatx8 acc[8];

  // layer 1 (K=288 padded)
  acc_zero8(acc);
  gemm_layer(aug + (size_t)(rowBase + m0 + lo) * 288, 288, W1, wsl, acc, tid, lane, hi, nh);
  epilogue_elu(acc, B1, act1, m0, lo, hi, nh);

  // layer 2
  acc_zero8(acc);
  gemm_layer(act1 + (m0 + lo) * 256, 256, W2, wsl, acc, tid, lane, hi, nh);
  epilogue_elu(acc, B2, act2, m0, lo, hi, nh);

  __syncthreads();
  // layer 3: 256 -> 1 dot, residual
  if (tid < 128) {
    size_t row = (size_t)rowBase + tid;
    float s = 0.f;
    for (int c = 0; c < 256; ++c) s += bf2f(act2[tid * 256 + c]) * W3[c];
    pred[row] = x[row] + s + B3[0];
  }
}

// ---------------------------------------------------------------------------
// Host orchestration
// ---------------------------------------------------------------------------
struct LinP { const float* w; const float* b; };
struct MlpP { LinP l1, l2, l3; const float* gamma; const float* beta; };

extern "C" void kernel_launch(void* const* d_in, const int* in_sizes, int n_in,
                              void* d_out, int out_size, void* d_ws, size_t ws_size,
                              hipStream_t stream) {
  (void)n_in; (void)out_size; (void)ws_size;
  auto F = [&](int i) { return (const float*)d_in[i]; };

  MlpP msg[4], dout, enc[4];
  LinP encOut, rg1, rg2, rinp, rout;
  const float* xin; const int* nsteps;

  if (in_sizes[0] == 4) {
    // jax pytree flatten: dict keys sorted alphabetically
    nsteps = (const int*)d_in[0];
    int i = 1;
    for (int k = 0; k < 4; ++k) {
      msg[k].l1.b = F(i++); msg[k].l1.w = F(i++);
      msg[k].l2.b = F(i++); msg[k].l2.w = F(i++);
      msg[k].l3.b = F(i++); msg[k].l3.w = F(i++);
    }
    dout.l1.b = F(i++); dout.l1.w = F(i++);
    dout.l2.b = F(i++); dout.l2.w = F(i++);
    dout.l3.b = F(i++); dout.l3.w = F(i++);
    for (int k = 0; k < 4; ++k) {
      enc[k].beta = F(i++); enc[k].gamma = F(i++);
      enc[k].l1.b = F(i++); enc[k].l1.w = F(i++);
      enc[k].l2.b = F(i++); enc[k].l2.w = F(i++);
      enc[k].l3.b = F(i++); enc[k].l3.w = F(i++);
    }
    encOut.b = F(i++); encOut.w = F(i++);
    rg1.b = F(i++); rg1.w = F(i++);
    rg2.b = F(i++); rg2.w = F(i++);
    rinp.b = F(i++); rinp.w = F(i++);
    rout.b = F(i++); rout.w = F(i++);
    xin = F(i++);
  } else {
    // insertion-order flatten
    int i = 0;
    xin = F(i++); nsteps = (const int*)d_in[i++];
    rinp.w = F(i++); rinp.b = F(i++);
    rg1.w = F(i++); rg1.b = F(i++);
    rg2.w = F(i++); rg2.b = F(i++);
    rout.w = F(i++); rout.b = F(i++);
    for (int k = 0; k < 4; ++k) {
      enc[k].l1.w = F(i++); enc[k].l1.b = F(i++);
      enc[k].l2.w = F(i++); enc[k].l2.b = F(i++);
      enc[k].l3.w = F(i++); enc[k].l3.b = F(i++);
      enc[k].gamma = F(i++); enc[k].beta = F(i++);
    }
    encOut.w = F(i++); encOut.b = F(i++);
    for (int k = 0; k < 4; ++k) {
      msg[k].l1.w = F(i++); msg[k].l1.b = F(i++);
      msg[k].l2.w = F(i++); msg[k].l2.b = F(i++);
      msg[k].l3.w = F(i++); msg[k].l3.b = F(i++);
    }
    dout.l1.w = F(i++); dout.l1.b = F(i++);
    dout.l2.w = F(i++); dout.l2.b = F(i++);
    dout.l3.w = F(i++); dout.l3.b = F(i++);
  }

  // ---- workspace carve (deterministic bump allocator, 256B aligned) ----
  char* p = (char*)d_ws;
  auto bump = [&](size_t bytes) -> char* {
    char* r = p; p += (bytes + 255) & ~(size_t)255; return r;
  };
  unsigned short* g1w  = (unsigned short*)bump((size_t)512 * 256 * 2);
  unsigned short* g2w  = (unsigned short*)bump((size_t)512 * 256 * 2);
  unsigned short* outw = (unsigned short*)bump((size_t)512 * 256 * 2);
  const int encK[4] = {256, 512, 256, 768};
  unsigned short *encw1[4], *encw2[4], *encw3[4];
  for (int k = 0; k < 4; ++k) {
    encw1[k] = (unsigned short*)bump((size_t)encK[k] * 256 * 2);
    encw2[k] = (unsigned short*)bump((size_t)256 * 256 * 2);
    encw3[k] = (unsigned short*)bump((size_t)256 * 256 * 2);
  }
  unsigned short *mw2[3], *mw3[3];
  for (int k = 0; k < 3; ++k) {
    mw2[k] = (unsigned short*)bump((size_t)256 * 256 * 2);
    mw3[k] = (unsigned short*)bump((size_t)256 * 256 * 2);
  }
  unsigned short* dw1 = (unsigned short*)bump((size_t)288 * 256 * 2);
  unsigned short* dw2 = (unsigned short*)bump((size_t)256 * 256 * 2);

  float* hA = (float*)bump((size_t)128 * 256 * 4);
  float* hB = (float*)bump((size_t)128 * 256 * 4);
  float* node_h = (float*)bump((size_t)128 * 256 * 4);
  unsigned short* A1 = (unsigned short*)bump((size_t)128 * 256 * 2);
  float* o1 = (float*)bump((size_t)128 * 256 * 4);
  unsigned short* z1 = (unsigned short*)bump((size_t)128 * 256 * 2);
  unsigned short* A2 = (unsigned short*)bump((size_t)3968 * 512 * 2);
  float* o2 = (float*)bump((size_t)3968 * 256 * 4);
  unsigned short* z2 = (unsigned short*)bump((size_t)3968 * 256 * 2);
  unsigned short* A3 = (unsigned short*)bump((size_t)128 * 256 * 2);
  float* o3 = (float*)bump((size_t)128 * 256 * 4);
  unsigned short* z3 = (unsigned short*)bump((size_t)128 * 256 * 2);
  unsigned short* A4 = (unsigned short*)bump((size_t)3968 * 768 * 2);
  float* o4 = (float*)bump((size_t)3968 * 256 * 4);
  unsigned short* z4 = (unsigned short*)bump((size_t)3968 * 256 * 2);
  float* edges = (float*)bump((size_t)3968 * 4 * 4);
  float* agg = (float*)bump((size_t)8192 * 256 * 4);
  unsigned short* aug = (unsigned short*)bump((size_t)8192 * 288 * 2);

  auto conv = [&](const float* src, unsigned short* dst, int srcRows, int dstRows) {
    int n = dstRows * 256;
    k_f2bf_pad<<<(n + 255) / 256, 256, 0, stream>>>(src, dst, srcRows, n);
  };

  // ---- weight conversion (f32 -> bf16, once per launch) ----
  conv(rg1.w, g1w, 512, 512);
  conv(rg2.w, g2w, 512, 512);
  conv(rout.w, outw, 512, 512);
  for (int k = 0; k < 4; ++k) {
    conv(enc[k].l1.w, encw1[k], encK[k], encK[k]);
    conv(enc[k].l2.w, encw2[k], 256, 256);
    conv(enc[k].l3.w, encw3[k], 256, 256);
  }
  for (int k = 0; k < 3; ++k) {
    conv(msg[k + 1].l2.w, mw2[k], 256, 256);
    conv(msg[k + 1].l3.w, mw3[k], 256, 256);
  }
  conv(dout.l1.w, dw1, 257, 288);   // pad K 257 -> 288
  conv(dout.l2.w, dw2, 256, 256);

  // ---- RNN over time (64 sequential steps, ping-pong h) ----
  k_zero_f32<<<128, 256, 0, stream>>>(hA, 128 * 256);
  float* cur = hA; float* nxt = hB;
  for (int t = 0; t < 64; ++t) {
    k_rnn_step<<<8, 256, 0, stream>>>(xin, t, cur, nxt, rinp.w, rinp.b,
                                      g1w, rg1.b, g2w, rg2.b, outw, rout.b,
                                      nsteps, node_h);
    float* tmp = cur; cur = nxt; nxt = tmp;
  }

  // ---- Encoder ----
  const size_t mlpLds = (size_t)2 * 128 * 256 * 2 + (size_t)32 * 256 * 2; // 144 KB
  conv(node_h, A1, 128, 128);
  k_fused_mlp3<<<1, 512, mlpLds, stream>>>(A1, 256, encw1[0], enc[0].l1.b,
      encw2[0], enc[0].l2.b, encw3[0], enc[0].l3.b, o1);
  k_bn<<<256, 256, 0, stream>>>(o1, 128, enc[0].gamma, enc[0].beta, z1);
  k_gather2<<<3968, 256, 0, stream>>>(z1, A2);
  k_fused_mlp3<<<31, 512, mlpLds, stream>>>(A2, 512, encw1[1], enc[1].l1.b,
      encw2[1], enc[1].l2.b, encw3[1], enc[1].l3.b, o2);
  k_bn<<<256, 256, 0, stream>>>(o2, 3968, enc[1].gamma, enc[1].beta, z2);
  k_edge2node<<<128, 256, 0, stream>>>(z2, A3);
  k_fused_mlp3<<<1, 512, mlpLds, stream>>>(A3, 256, encw1[2], enc[2].l1.b,
      encw2[2], enc[2].l2.b, encw3[2], enc[2].l3.b, o3);
  k_bn<<<256, 256, 0, stream>>>(o3, 128, enc[2].gamma, enc[2].beta, z3);
  k_gather3<<<3968, 256, 0, stream>>>(z3, z2, A4);
  k_fused_mlp3<<<31, 512, mlpLds, stream>>>(A4, 768, encw1[3], enc[3].l1.b,
      encw2[3], enc[3].l2.b, encw3[3], enc[3].l3.b, o4);
  k_bn<<<256, 256, 0, stream>>>(o4, 3968, enc[3].gamma, enc[3].beta, z4);
  k_enc_out<<<16, 256, 0, stream>>>(z4, encOut.w, encOut.b, edges);

  // ---- Decoder ----
  MsgW mwp[3];
  for (int k = 0; k < 3; ++k)
    mwp[k] = { msg[k + 1].l1.w, msg[k + 1].l1.b, mw2[k], msg[k + 1].l2.b,
               mw3[k], msg[k + 1].l3.b };
  const size_t decMsgLds = (size_t)2 * 128 * 256 * 2   // act1 + act2
                         + (size_t)32 * 256 * 2        // weight slice staging
                         + (size_t)128 * 256 * 4       // f32 accumulator
                         + (size_t)(128 + 128 + 3 * 128) * 4;
  k_dec_msg<<<2048, 512, decMsgLds, stream>>>(xin, edges, mwp[0], mwp[1], mwp[2], agg);
  k_build_aug<<<8192, 256, 0, stream>>>(xin, agg, aug);
  k_dec_out<<<64, 512, mlpLds, stream>>>(aug, dw1, dout.l1.b, dw2, dout.l2.b,
                                         dout.l3.w, dout.l3.b, xin, (float*)d_out);
}
